// DHCNN_FAM_ViM_Block_69870527972112
// MI455X (gfx1250) — matrix operand order repaired and checked by reading, not compiled, not run on hardware
//
#include <hip/hip_runtime.h>

#define DEV __device__ __forceinline__

typedef __attribute__((ext_vector_type(16))) __bf16 bf16x16;
typedef __attribute__((ext_vector_type(8)))  float  f32x8;

union FragH { unsigned int u[8]; bf16x16 v; };

DEV unsigned short f2bf(float f) {
  unsigned int u = __float_as_uint(f);
  u += 0x7FFFu + ((u >> 16) & 1u);
  return (unsigned short)(u >> 16);
}
DEV unsigned int packbf(float f0, float f1) {
  return (unsigned int)f2bf(f0) | ((unsigned int)f2bf(f1) << 16);
}
DEV float gelu_f(float x) { return 0.5f * x * (1.0f + erff(x * 0.70710678118654752f)); }
DEV float sigm(float x) { return 1.0f / (1.0f + expf(-x)); }

// Contiguous-K A fragment: lane needs K = khalf*8+0..7 and 16+khalf*8+0..7
// from a 16B-aligned f32 row -> four b128 loads, packed to bf16 pairs.
DEV void load_a_contig(const float* __restrict__ p, int khalf, FragH& a) {
  const float4* q = (const float4*)(p + (khalf << 3));
  float4 x0 = q[0];
  float4 x1 = q[1];
  const float4* r = (const float4*)(p + 16 + (khalf << 3));
  float4 y0 = r[0];
  float4 y1 = r[1];
  a.u[0] = packbf(x0.x, x0.y); a.u[1] = packbf(x0.z, x0.w);
  a.u[2] = packbf(x1.x, x1.y); a.u[3] = packbf(x1.z, x1.w);
  a.u[4] = packbf(y0.x, y0.y); a.u[5] = packbf(y0.z, y0.w);
  a.u[6] = packbf(y1.x, y1.y); a.u[7] = packbf(y1.z, y1.w);
}

// ---------------------------------------------------------------------------
// Generic bf16 WMMA GEMM: C[M,N] = A[M,K] * B[K,N] (+ epilogue)
// Wave tile: 32(M) x 64(N) = 2 A-fragments x 4 B-fragments -> 8 WMMA / chunk.
// AMODE 0: A affine  addr = A + batch*sAb + m*sAm + k*sAk   (CONTIG if sAk==1)
// AMODE 1: A im2col over sequence: row m=(b,l), K = tap*rowLen + kk (CONTIG)
// AMODE 2: A = NHWC image rows: m=(b,c,h), k=w  (hardcoded 8x64x224x224)
// B pre-packed into WMMA fragment order (see prepack_b_kernel).
// epi: 0 raw | 1 gelu(bn) | 3 +bias | 5 *sigmoid(G) | 6 +bias+R | 7 accumulate
// ---------------------------------------------------------------------------
template <int AMODE, bool CONTIG>
__global__ void __launch_bounds__(256) gemm_bf16_wmma(
    const float* __restrict__ A, const unsigned int* __restrict__ Bf,
    float* __restrict__ C, int M, int N, int K,
    long sAm, long sAk, long sAb,
    int rowLen, int Lseq,
    long ldc, long sCb,
    int epi,
    const float* __restrict__ e0, const float* __restrict__ e1,
    const float* __restrict__ R, long ldr,
    const float* __restrict__ G, long ldg, int goff) {
  int lane = threadIdx.x & 31;
  int wid  = blockIdx.x * (blockDim.x >> 5) + (threadIdx.x >> 5);
  int mtiles = M >> 5;   // 32 rows per wave tile
  int ntiles = N >> 4;
  int nquads = ntiles >> 2;
  if (wid >= mtiles * nquads) return;
  int mt = wid % mtiles;
  int nq = wid / mtiles;
  long batch = blockIdx.y;
  const float* Ab = A + batch * sAb;
  float*       Cb = C + batch * sCb;

  int khalf = lane >> 4;
  int mrow[2];
  mrow[0] = (mt << 5) + (lane & 15);
  mrow[1] = mrow[0] + 16;

  const float* Abase[2] = {A, A};
  long stepk = sAk;
  int bb[2] = {0, 0}, ll[2] = {0, 0};
#pragma unroll
  for (int s = 0; s < 2; ++s) {
    if (AMODE == 0) {
      Abase[s] = Ab + (long)mrow[s] * sAm;
    } else if (AMODE == 2) {
      int h  = mrow[s] % 224;
      int bc = mrow[s] / 224;
      int b = bc >> 6, c = bc & 63;
      Abase[s] = Ab + (long)b * 3211264 + (long)h * 14336 + c;
    } else {  // AMODE 1
      bb[s] = mrow[s] / Lseq;
      ll[s] = mrow[s] - bb[s] * Lseq;
    }
  }
  if (AMODE == 2) stepk = 64;

  f32x8 acc[2][4] = {};
  int nk = K >> 5;
  for (int kt = 0; kt < nk; ++kt) {
    FragH a[2];
    if (AMODE == 1) {
      int kk0 = kt << 5;
      int tap = kk0 / rowLen;
      int kin = kk0 - tap * rowLen;
#pragma unroll
      for (int s = 0; s < 2; ++s) {
        int lsrc = ll[s] + tap - 1;
        if (lsrc >= 0 && lsrc < Lseq) {
          const float* Ap = Ab + ((long)(bb[s] * Lseq + lsrc) * rowLen + kin);
          load_a_contig(Ap, khalf, a[s]);
        } else {
#pragma unroll
          for (int v = 0; v < 8; ++v) a[s].u[v] = 0u;
        }
      }
    } else if (CONTIG) {
      long k0 = (long)kt << 5;
#pragma unroll
      for (int s = 0; s < 2; ++s) load_a_contig(Abase[s] + k0, khalf, a[s]);
      if (kt + 1 < nk) {
        __builtin_prefetch(Abase[0] + k0 + 32, 0, 1);
        __builtin_prefetch(Abase[1] + k0 + 32, 0, 1);
      }
    } else {
      long k0 = (long)kt << 5;
#pragma unroll
      for (int s = 0; s < 2; ++s) {
#pragma unroll
        for (int v = 0; v < 8; ++v) {
          int kloc = ((v < 4) ? 0 : 16) + (khalf << 3) + ((v & 3) << 1);
          float f0 = Abase[s][(k0 + kloc) * stepk];
          float f1 = Abase[s][(k0 + kloc + 1) * stepk];
          a[s].u[v] = packbf(f0, f1);
        }
      }
      if (kt + 1 < nk) __builtin_prefetch(Abase[0] + (k0 + 32) * stepk, 0, 1);
    }
#pragma unroll
    for (int nt = 0; nt < 4; ++nt) {
      int ntab = (nq << 2) + nt;
      const uint4* bp = (const uint4*)(Bf + ((((long)kt * ntiles + ntab) << 5) + lane) * 8);
      FragH b;
      uint4 q0 = bp[0], q1 = bp[1];
      b.u[0] = q0.x; b.u[1] = q0.y; b.u[2] = q0.z; b.u[3] = q0.w;
      b.u[4] = q1.x; b.u[5] = q1.y; b.u[6] = q1.z; b.u[7] = q1.w;
      acc[0][nt] = __builtin_amdgcn_wmma_f32_16x16x32_bf16(
          false, a[0].v, false, b.v, (short)0, acc[0][nt], false, false);
      acc[1][nt] = __builtin_amdgcn_wmma_f32_16x16x32_bf16(
          false, a[1].v, false, b.v, (short)0, acc[1][nt], false, false);
    }
  }

  int ncol0 = (nq << 6) + (lane & 15);
#pragma unroll
  for (int s = 0; s < 2; ++s) {
    int mbase = (mt << 5) + (s << 4) + (khalf << 3);
#pragma unroll
    for (int nt = 0; nt < 4; ++nt) {
      int n = ncol0 + (nt << 4);
#pragma unroll
      for (int r = 0; r < 8; ++r) {
        int m = mbase + r;
        float v = acc[s][nt][r];
        float o;
        if      (epi == 0) o = v;
        else if (epi == 1) o = gelu_f(v * e0[n] + e1[n]);
        else if (epi == 3) o = v + e0[n];
        else if (epi == 5) o = v * sigm(G[(long)m * ldg + goff + n]);
        else if (epi == 6) o = v + e0[n] + R[(long)m * ldr + n];
        else               o = v + Cb[(long)m * ldc + n];  // 7: accumulate
        Cb[(long)m * ldc + n] = o;
      }
    }
  }
}

// -------------------- weight fragment packing ------------------------------
__global__ void prepack_b_kernel(const unsigned short* __restrict__ Braw,
                                 unsigned int* __restrict__ Bf, int K, int N) {
  long t = (long)blockIdx.x * blockDim.x + threadIdx.x;
  long total = ((long)K * N) >> 1;
  if (t >= total) return;
  int v = (int)(t & 7);
  int lane = (int)((t >> 3) & 31);
  long rest = t >> 8;
  int ntiles = N >> 4;
  int nt = (int)(rest % ntiles);
  int kt = (int)(rest / ntiles);
  int n = (nt << 4) + (lane & 15);
  int k = (kt << 5) + ((v & 4) << 2) + ((lane >> 4) << 3) + ((v & 3) << 1);
  unsigned int lo = Braw[(long)k * N + n];
  unsigned int hi = Braw[(long)(k + 1) * N + n];
  Bf[t] = lo | (hi << 16);
}

// Braw[k][n] = w[n*ldw + koff + k]
__global__ void fill_bT(const float* __restrict__ w, unsigned short* __restrict__ Braw,
                        int K, int N, int ldw, int koff) {
  long t = (long)blockIdx.x * blockDim.x + threadIdx.x;
  if (t >= (long)K * N) return;
  int n = (int)(t % N);
  int k = (int)(t / N);
  Braw[t] = f2bf(w[(long)n * ldw + koff + k]);
}

// N = 2*Nh, Braw[k][n] = (n<Nh? wf : wb)[(n%Nh)*K + k]
__global__ void fill_bT2(const float* __restrict__ wf, const float* __restrict__ wb,
                         unsigned short* __restrict__ Braw, int K, int Nh) {
  int N = 2 * Nh;
  long t = (long)blockIdx.x * blockDim.x + threadIdx.x;
  if (t >= (long)K * N) return;
  int n = (int)(t % N);
  int k = (int)(t / N);
  const float* w = (n < Nh) ? wf : wb;
  Braw[t] = f2bf(w[(long)(n % Nh) * K + k]);
}

// K=128, N=256 : k<64 -> wf[n*64+k] else wb[n*64+(k-64)]   (w_out stacking)
__global__ void fill_bwout(const float* __restrict__ wf, const float* __restrict__ wb,
                           unsigned short* __restrict__ Braw) {
  long t = (long)blockIdx.x * blockDim.x + threadIdx.x;
  if (t >= 128L * 256L) return;
  int n = (int)(t % 256);
  int k = (int)(t / 256);
  float val = (k < 64) ? wf[n * 64 + k] : wb[n * 64 + (k - 64)];
  Braw[t] = f2bf(val);
}

// seq conv weights (Nout, rowLen, 3): K = 3*rowLen, B[tap*rowLen+r][n]=w[(n*rowLen+r)*3+tap]
__global__ void fill_bseq(const float* __restrict__ w, unsigned short* __restrict__ Braw,
                          int rowLen, int N) {
  int K = 3 * rowLen;
  long t = (long)blockIdx.x * blockDim.x + threadIdx.x;
  if (t >= (long)K * N) return;
  int n = (int)(t % N);
  int k = (int)(t / N);
  int tap = k / rowLen;
  int r = k - tap * rowLen;
  Braw[t] = f2bf(w[((long)n * rowLen + r) * 3 + tap]);
}

// Row DFT twiddles: K=224 (w), N=256 : n<113 cos(2pi*k*n/224); n in [128,241) -> -sin
__global__ void fill_twrow(unsigned short* __restrict__ Braw) {
  long t = (long)blockIdx.x * blockDim.x + threadIdx.x;
  if (t >= 224L * 256L) return;
  int n = (int)(t & 255);
  int k = (int)(t >> 8);
  float val = 0.0f;
  if (n < 113) {
    float ang = 6.283185307179586f * (float)((k * n) % 224) / 224.0f;
    val = cosf(ang);
  } else if (n >= 128 && n < 241) {
    int vv = n - 128;
    float ang = 6.283185307179586f * (float)((k * vv) % 224) / 224.0f;
    val = -sinf(ang);
  }
  Braw[t] = f2bf(val);
}

// Column DFT twiddles: K=224 (h), N=448 (u | 224+u)
// part0 (applied to Xr): [h][u]=cos, [h][224+u]=-sin
// part1 (applied to Xi): [h][u]=sin, [h][224+u]= cos
__global__ void fill_twcol(unsigned short* __restrict__ Braw, int part) {
  long t = (long)blockIdx.x * blockDim.x + threadIdx.x;
  if (t >= 224L * 448L) return;
  int n = (int)(t % 448);
  int h = (int)(t / 448);
  int u = (n < 224) ? n : (n - 224);
  float ang = 6.283185307179586f * (float)((u * h) % 224) / 224.0f;
  float val;
  if (part == 0) val = (n < 224) ? cosf(ang) : -sinf(ang);
  else           val = (n < 224) ? sinf(ang) :  cosf(ang);
  Braw[t] = f2bf(val);
}

// -------------------- elementwise / misc kernels ---------------------------
#define ELEM_IMG 25690112L  // 8*64*224*224

__global__ void bn_prep(const float* g, const float* b, const float* m, const float* v,
                        float* scale, float* shift) {
  int i = threadIdx.x;
  if (i < 64) {
    float s = g[i] * rsqrtf(v[i] + 1e-5f);
    scale[i] = s;
    shift[i] = b[i] - m[i] * s;
  }
}

__global__ void nchw2nhwc(const float* __restrict__ in, float* __restrict__ out) {
  long i = (long)blockIdx.x * blockDim.x + threadIdx.x;
  if (i >= ELEM_IMG) return;
  int c = (int)(i & 63);
  long p = i >> 6;
  int w = (int)(p % 224);
  long p2 = p / 224;
  int h = (int)(p2 % 224);
  int b = (int)(p2 / 224);
  out[i] = in[(((long)b * 64 + c) * 224 + h) * 224 + w];
}

__global__ void dwconv2d(const float* __restrict__ h1, const float* __restrict__ wgt,
                         const float* __restrict__ scale, const float* __restrict__ shift,
                         float* __restrict__ out, int dil) {
  long i = (long)blockIdx.x * blockDim.x + threadIdx.x;
  if (i >= ELEM_IMG) return;
  int c = (int)(i & 63);
  long p = i >> 6;
  int w = (int)(p % 224);
  long p2 = p / 224;
  int h = (int)(p2 % 224);
  int b = (int)(p2 / 224);
  float acc = 0.0f;
#pragma unroll
  for (int ty = 0; ty < 3; ++ty) {
    int hh = h + (ty - 1) * dil;
    if (hh < 0 || hh >= 224) continue;
#pragma unroll
    for (int tx = 0; tx < 3; ++tx) {
      int ww = w + (tx - 1) * dil;
      if (ww < 0 || ww >= 224) continue;
      acc += h1[(((long)b * 224 + hh) * 224 + ww) * 64 + c] * wgt[c * 9 + ty * 3 + tx];
    }
  }
  out[i] = gelu_f(acc * scale[c] + shift[c]);
}

__global__ void fuse_final(const float* __restrict__ acc, const float* __restrict__ xin,
                           const float* __restrict__ scale, const float* __restrict__ shift,
                           float* __restrict__ x1) {
  long i = (long)blockIdx.x * blockDim.x + threadIdx.x;
  if (i >= ELEM_IMG) return;
  int c = (int)(i & 63);
  x1[i] = xin[i] + gelu_f(acc[i] * scale[c] + shift[c]);
}

__global__ void fam_reduce(const float* __restrict__ S, float* __restrict__ desc) {
  int bc = blockIdx.x;
  const float* p = S + (long)bc * 57344;
  float s = 0.0f;
  for (int i = threadIdx.x; i < 25312; i += blockDim.x) {  // 113*224
    int v = i / 224;
    int u = i - v * 224;
    float re = p[v * 448 + u];
    float im = p[v * 448 + 224 + u];
    s += sqrtf(re * re + im * im);
  }
  __shared__ float red[256];
  red[threadIdx.x] = s;
  __syncthreads();
  for (int st = 128; st > 0; st >>= 1) {
    if (threadIdx.x < st) red[threadIdx.x] += red[threadIdx.x + st];
    __syncthreads();
  }
  if (threadIdx.x == 0) desc[bc] = red[0] * (1.0f / 5669888.0f);  // /(224*224*113)
}

__global__ void fam_mlp(const float* __restrict__ desc,
                        const float* __restrict__ w1, const float* __restrict__ b1,
                        const float* __restrict__ w2, const float* __restrict__ b2,
                        float* __restrict__ wch) {
  __shared__ float sd[512];
  __shared__ float sh[64];
  int t = threadIdx.x;
  sd[t] = desc[t];
  __syncthreads();
  if (t < 64) {
    int b = t >> 3, j = t & 7;
    float a = b1[j];
    for (int c = 0; c < 64; ++c) a += sd[b * 64 + c] * w1[j * 64 + c];
    sh[t] = gelu_f(a);
  }
  __syncthreads();
  {
    int b = t >> 6, c = t & 63;
    float a = b2[c];
    for (int j = 0; j < 8; ++j) a += sh[b * 8 + j] * w2[c * 8 + j];
    wch[t] = sigm(a);
  }
}

__global__ void patchify_scale(const float* __restrict__ x1, const float* __restrict__ wch,
                               float* __restrict__ t) {
  long i = (long)blockIdx.x * blockDim.x + threadIdx.x;
  if (i >= ELEM_IMG) return;
  int c = (int)(i & 63);
  long p = i >> 6;
  int pq = (int)(p % 49);
  long p2 = p / 49;
  int l = (int)(p2 % 1024);
  int b = (int)(p2 / 1024);
  int pp = pq / 7, qq = pq - pp * 7;
  int hp = l >> 5, wp = l & 31;
  int h = hp * 7 + pp, w = wp * 7 + qq;
  t[i] = x1[(((long)b * 224 + h) * 224 + w) * 64 + c] * wch[b * 64 + c];
}

__global__ void rmsnorm_k(const float* __restrict__ s, const float* __restrict__ g,
                          float* __restrict__ Tn) {
  int row = blockIdx.x;
  int t = threadIdx.x;
  float x = s[(long)row * 256 + t];
  __shared__ float red[256];
  red[t] = x * x;
  __syncthreads();
  for (int st = 128; st > 0; st >>= 1) {
    if (t < st) red[t] += red[t + st];
    __syncthreads();
  }
  float rms = sqrtf(red[0] * (1.0f / 256.0f) + 1e-6f);
  Tn[(long)row * 256 + t] = x / rms * g[t];
}

__global__ void convdw1d(const float* __restrict__ xz, const float* __restrict__ w,
                         const float* __restrict__ bias, float* __restrict__ out) {
  long i = (long)blockIdx.x * blockDim.x + threadIdx.x;
  if (i >= 8192L * 256L) return;
  int d = (int)(i & 255);
  long m = i >> 8;
  int l = (int)(m & 1023);
  long b = m >> 10;
  float acc = bias[d];
#pragma unroll
  for (int tap = 0; tap < 3; ++tap) {
    int ls = l + tap - 1;
    if (ls >= 0 && ls < 1024) acc += xz[((b * 1024 + ls) * 512) + d] * w[d * 3 + tap];
  }
  out[i] = acc;
}

__global__ void ssm_scan(const float* __restrict__ u, float* __restrict__ ys,
                         const float* lAf, const float* Bfp, const float* Cfp, const float* dtf,
                         const float* lAb, const float* Bbp, const float* Cbp, const float* dtb) {
  int t = blockIdx.x * blockDim.x + threadIdx.x;
  if (t >= 1024) return;
  int dir = t >> 9;
  int r = t & 511;
  int b = r >> 6;
  int n = r & 63;
  float dt   = expf(dir ? dtb[0] : dtf[0]);
  float Aa   = -expf(dir ? lAb[n] : lAf[n]);
  float Abar = expf(Aa * dt);
  float frac = (fabsf(Aa) < 1e-6f) ? dt : (Abar - 1.0f) / Aa;
  float Bbar = frac * (dir ? Bbp[n] : Bfp[n]);
  float Cn   = dir ? Cbp[n] : Cfp[n];
  float h = 0.0f;
  long base = (long)b * 1024 * 128 + (dir ? 64 : 0) + n;
  if (dir == 0) {
    for (int l = 0; l < 1024; ++l) {
      long o = base + (long)l * 128;
      h = h * Abar + u[o] * Bbar;
      ys[o] = h * Cn;
    }
  } else {
    for (int l = 1023; l >= 0; --l) {
      long o = base + (long)l * 128;
      h = h * Abar + u[o] * Bbar;
      ys[o] = h * Cn;
    }
  }
}

__global__ void unpatchify(const float* __restrict__ tk, float* __restrict__ out) {
  long i = (long)blockIdx.x * blockDim.x + threadIdx.x;
  if (i >= ELEM_IMG) return;
  int w = (int)(i % 224);
  long p = i / 224;
  int h = (int)(p % 224);
  long p2 = p / 224;
  int c = (int)(p2 & 63);
  int b = (int)(p2 >> 6);
  int hp = h / 7, pp = h - hp * 7;
  int wp = w / 7, qq = w - wp * 7;
  out[i] = tk[(((long)b * 1024 + hp * 32 + wp) * 3136) + (pp * 7 + qq) * 64 + c];
}

// ---------------------------------------------------------------------------
// host side
// ---------------------------------------------------------------------------
static inline void run_gemm(int amode, const float* A, const unsigned int* Bf, float* C,
                            int M, int N, int K, long sAm, long sAk, long sAb,
                            int rowLen, int Lseq, long ldc, long sCb, int batches, int epi,
                            const float* e0, const float* e1, const float* R, long ldr,
                            const float* G, long ldg, int goff, hipStream_t stream) {
  long waves = (long)(M >> 5) * (N >> 6);
  dim3 grid((unsigned)((waves + 7) / 8), (unsigned)batches, 1);
  dim3 block(256, 1, 1);
  if (amode == 0 && sAk == 1)
    gemm_bf16_wmma<0, true><<<grid, block, 0, stream>>>(A, Bf, C, M, N, K, sAm, sAk, sAb,
        rowLen, Lseq, ldc, sCb, epi, e0, e1, R, ldr, G, ldg, goff);
  else if (amode == 0)
    gemm_bf16_wmma<0, false><<<grid, block, 0, stream>>>(A, Bf, C, M, N, K, sAm, sAk, sAb,
        rowLen, Lseq, ldc, sCb, epi, e0, e1, R, ldr, G, ldg, goff);
  else if (amode == 1)
    gemm_bf16_wmma<1, true><<<grid, block, 0, stream>>>(A, Bf, C, M, N, K, sAm, sAk, sAb,
        rowLen, Lseq, ldc, sCb, epi, e0, e1, R, ldr, G, ldg, goff);
  else
    gemm_bf16_wmma<2, false><<<grid, block, 0, stream>>>(A, Bf, C, M, N, K, sAm, sAk, sAb,
        rowLen, Lseq, ldc, sCb, epi, e0, e1, R, ldr, G, ldg, goff);
}

static inline void prep_pack(const unsigned short* Braw, unsigned int* Bf, int K, int N,
                             hipStream_t stream) {
  long dwords = ((long)K * N) >> 1;
  prepack_b_kernel<<<(unsigned)((dwords + 255) / 256), 256, 0, stream>>>(Braw, Bf, K, N);
}

extern "C" void kernel_launch(void* const* d_in, const int* in_sizes, int n_in,
                              void* d_out, int out_size, void* d_ws, size_t ws_size,
                              hipStream_t stream) {
  (void)in_sizes; (void)n_in; (void)out_size; (void)ws_size;
  const float* x        = (const float*)d_in[0];
  const float* pw_in_w  = (const float*)d_in[1];
  const float* bn_in[4]   = {(const float*)d_in[2], (const float*)d_in[3], (const float*)d_in[4], (const float*)d_in[5]};
  const float* dw_w[3]    = {(const float*)d_in[6], (const float*)d_in[11], (const float*)d_in[16]};
  const float* bn_d[3][4] = {
      {(const float*)d_in[7],  (const float*)d_in[8],  (const float*)d_in[9],  (const float*)d_in[10]},
      {(const float*)d_in[12], (const float*)d_in[13], (const float*)d_in[14], (const float*)d_in[15]},
      {(const float*)d_in[17], (const float*)d_in[18], (const float*)d_in[19], (const float*)d_in[20]}};
  const float* fuse_w   = (const float*)d_in[21];
  const float* bn_f[4]  = {(const float*)d_in[22], (const float*)d_in[23], (const float*)d_in[24], (const float*)d_in[25]};
  const float* fam_w1   = (const float*)d_in[26];
  const float* fam_b1   = (const float*)d_in[27];
  const float* fam_w2   = (const float*)d_in[28];
  const float* fam_b2   = (const float*)d_in[29];
  const float* seq_in_w = (const float*)d_in[30];
  const float* seq_in_b = (const float*)d_in[31];
  const float* rms_g    = (const float*)d_in[32];
  const float* in_lin_w = (const float*)d_in[33];
  const float* in_lin_b = (const float*)d_in[34];
  const float* cdw_w    = (const float*)d_in[35];
  const float* cdw_b    = (const float*)d_in[36];
  const float* cpw_w    = (const float*)d_in[37];
  const float* cpw_b    = (const float*)d_in[38];
  const float* sf_win   = (const float*)d_in[39];
  const float* sf_wout  = (const float*)d_in[40];
  const float* sf_lA    = (const float*)d_in[41];
  const float* sf_B     = (const float*)d_in[42];
  const float* sf_C     = (const float*)d_in[43];
  const float* sf_dt    = (const float*)d_in[44];
  const float* sb_win   = (const float*)d_in[45];
  const float* sb_wout  = (const float*)d_in[46];
  const float* sb_lA    = (const float*)d_in[47];
  const float* sb_B     = (const float*)d_in[48];
  const float* sb_C     = (const float*)d_in[49];
  const float* sb_dt    = (const float*)d_in[50];
  const float* ol_w     = (const float*)d_in[51];
  const float* ol_b     = (const float*)d_in[52];
  const float* seq_out_w = (const float*)d_in[53];
  const float* seq_out_b = (const float*)d_in[54];

  // ---- workspace carve-up ----
  char* Wp = (char*)d_ws;
  size_t off = 0;
  auto alloc = [&](size_t bytes) -> void* {
    off = (off + 255) & ~(size_t)255;
    void* p = Wp + off;
    off += bytes;
    return p;
  };
  float* buf0 = (float*)alloc(117440512);  // x_nhwc -> colOut
  float* buf1 = (float*)alloc(102760448);  // h1 -> x1
  float* buf2 = (float*)alloc(117440512);  // tmp64 -> RI -> tk
  float* buf3 = (float*)alloc(102760448);  // fuse acc -> t
  float* s0   = (float*)alloc(8388608);
  float* Tn   = (float*)alloc(8388608);
  float* xz   = (float*)alloc(16777216);
  float* xc1  = (float*)alloc(8388608);
  float* xc   = (float*)alloc(8388608);
  float* ubuf = (float*)alloc(4194304);
  float* ysb  = (float*)alloc(4194304);
  float* yg   = (float*)alloc(8388608);
  float* s1   = (float*)alloc(8388608);
  float* desc = (float*)alloc(2048);
  float* wch  = (float*)alloc(2048);
  float* sc_in = (float*)alloc(256); float* sh_in = (float*)alloc(256);
  float* sc_d[3]; float* sh_d[3];
  for (int i = 0; i < 3; ++i) { sc_d[i] = (float*)alloc(256); sh_d[i] = (float*)alloc(256); }
  float* sc_f = (float*)alloc(256); float* sh_f = (float*)alloc(256);
  unsigned short* Braw = (unsigned short*)alloc(4816896);
  unsigned int* F_pw     = (unsigned int*)alloc(8192);
  unsigned int* F_fuse   = (unsigned int*)alloc(8192);
  unsigned int* F_rowT   = (unsigned int*)alloc(114688);
  unsigned int* F_colT   = (unsigned int*)alloc(200704);
  unsigned int* F_colB   = (unsigned int*)alloc(200704);
  unsigned int* F_seqin  = (unsigned int*)alloc(4816896);
  unsigned int* F_inlin  = (unsigned int*)alloc(262144);
  unsigned int* F_cpw    = (unsigned int*)alloc(131072);
  unsigned int* F_u      = (unsigned int*)alloc(65536);
  unsigned int* F_y      = (unsigned int*)alloc(65536);
  unsigned int* F_ol     = (unsigned int*)alloc(131072);
  unsigned int* F_seqout = (unsigned int*)alloc(4816896);

  const unsigned EWG = (unsigned)((ELEM_IMG + 255) / 256);
  auto g1 = [](long n) { return (unsigned)((n + 255) / 256); };

  // ---- BN constants ----
  bn_prep<<<1, 64, 0, stream>>>(bn_in[0], bn_in[1], bn_in[2], bn_in[3], sc_in, sh_in);
  for (int i = 0; i < 3; ++i)
    bn_prep<<<1, 64, 0, stream>>>(bn_d[i][0], bn_d[i][1], bn_d[i][2], bn_d[i][3], sc_d[i], sh_d[i]);
  bn_prep<<<1, 64, 0, stream>>>(bn_f[0], bn_f[1], bn_f[2], bn_f[3], sc_f, sh_f);

  // ---- weight fragment packing (Braw scratch reused; stream-ordered) ----
  fill_bT<<<g1(64L * 64), 256, 0, stream>>>(pw_in_w, Braw, 64, 64, 64, 0);
  prep_pack(Braw, F_pw, 64, 64, stream);
  fill_twrow<<<g1(224L * 256), 256, 0, stream>>>(Braw);
  prep_pack(Braw, F_rowT, 224, 256, stream);
  fill_twcol<<<g1(224L * 448), 256, 0, stream>>>(Braw, 0);
  prep_pack(Braw, F_colT, 224, 448, stream);
  fill_twcol<<<g1(224L * 448), 256, 0, stream>>>(Braw, 1);
  prep_pack(Braw, F_colB, 224, 448, stream);
  fill_bseq<<<g1(9408L * 256), 256, 0, stream>>>(seq_in_w, Braw, 3136, 256);
  prep_pack(Braw, F_seqin, 9408, 256, stream);
  fill_bT<<<g1(256L * 512), 256, 0, stream>>>(in_lin_w, Braw, 256, 512, 256, 0);
  prep_pack(Braw, F_inlin, 256, 512, stream);
  fill_bT<<<g1(256L * 256), 256, 0, stream>>>(cpw_w, Braw, 256, 256, 256, 0);
  prep_pack(Braw, F_cpw, 256, 256, stream);
  fill_bT2<<<g1(256L * 128), 256, 0, stream>>>(sf_win, sb_win, Braw, 256, 64);
  prep_pack(Braw, F_u, 256, 128, stream);
  fill_bwout<<<g1(128L * 256), 256, 0, stream>>>(sf_wout, sb_wout, Braw);
  prep_pack(Braw, F_y, 128, 256, stream);
  fill_bT<<<g1(256L * 256), 256, 0, stream>>>(ol_w, Braw, 256, 256, 256, 0);
  prep_pack(Braw, F_ol, 256, 256, stream);
  fill_bseq<<<g1(768L * 3136), 256, 0, stream>>>(seq_out_w, Braw, 256, 3136);
  prep_pack(Braw, F_seqout, 768, 3136, stream);

  // ---- Phase 1: layout + pw_in(BN,GELU) ----
  nchw2nhwc<<<EWG, 256, 0, stream>>>(x, buf0);
  run_gemm(0, buf0, F_pw, buf1, 401408, 64, 64, 64, 1, 0, 0, 0, 64, 0, 1,
           /*epi*/1, sc_in, sh_in, nullptr, 0, nullptr, 0, 0, stream);

  // ---- Phase 2: dilated depthwise branches + fused 1x1 (accumulated) ----
  for (int i = 0; i < 3; ++i) {
    dwconv2d<<<EWG, 256, 0, stream>>>(buf1, dw_w[i], sc_d[i], sh_d[i], buf2, i + 1);
    fill_bT<<<g1(64L * 64), 256, 0, stream>>>(fuse_w, Braw, 64, 64, 192, i * 64);
    prep_pack(Braw, F_fuse, 64, 64, stream);
    run_gemm(0, buf2, F_fuse, buf3, 401408, 64, 64, 64, 1, 0, 0, 0, 64, 0, 1,
             /*epi*/(i == 0 ? 0 : 7), nullptr, nullptr, nullptr, 0, nullptr, 0, 0, stream);
  }
  fuse_final<<<EWG, 256, 0, stream>>>(buf3, buf0, sc_f, sh_f, buf1);  // x1 -> buf1

  // ---- Phase 3: FAM via DFT GEMMs ----
  run_gemm(2, buf1, F_rowT, buf2, 114688, 256, 224, 0, 0, 0, 0, 0, 256, 0, 1,
           0, nullptr, nullptr, nullptr, 0, nullptr, 0, 0, stream);           // RI
  run_gemm(0, buf2, F_colT, buf0, 128, 448, 224, 1, 256, 57344, 0, 0, 448, 57344, 512,
           0, nullptr, nullptr, nullptr, 0, nullptr, 0, 0, stream);           // Re/Im from Xr
  run_gemm(0, buf2 + 128, F_colB, buf0, 128, 448, 224, 1, 256, 57344, 0, 0, 448, 57344, 512,
           7, nullptr, nullptr, nullptr, 0, nullptr, 0, 0, stream);           // += from Xi
  fam_reduce<<<512, 256, 0, stream>>>(buf0, desc);
  fam_mlp<<<1, 512, 0, stream>>>(desc, fam_w1, fam_b1, fam_w2, fam_b2, wch);

  // ---- Phase 4: patchify with channel gating ----
  patchify_scale<<<EWG, 256, 0, stream>>>(buf1, wch, buf3);  // t -> buf3

  // ---- Phase 5: seq_in (im2col conv1d as GEMM) ----
  run_gemm(1, buf3, F_seqin, s0, 8192, 256, 9408, 0, 0, 0, 3136, 1024, 256, 0, 1,
           3, seq_in_b, nullptr, nullptr, 0, nullptr, 0, 0, stream);

  // ---- Phase 6: ViM block ----
  rmsnorm_k<<<8192, 256, 0, stream>>>(s0, rms_g, Tn);
  run_gemm(0, Tn, F_inlin, xz, 8192, 512, 256, 256, 1, 0, 0, 0, 512, 0, 1,
           3, in_lin_b, nullptr, nullptr, 0, nullptr, 0, 0, stream);
  convdw1d<<<8192, 256, 0, stream>>>(xz, cdw_w, cdw_b, xc1);
  run_gemm(0, xc1, F_cpw, xc, 8192, 256, 256, 256, 1, 0, 0, 0, 256, 0, 1,
           3, cpw_b, nullptr, nullptr, 0, nullptr, 0, 0, stream);
  run_gemm(0, xc, F_u, ubuf, 8192, 128, 256, 256, 1, 0, 0, 0, 128, 0, 1,
           0, nullptr, nullptr, nullptr, 0, nullptr, 0, 0, stream);
  ssm_scan<<<4, 256, 0, stream>>>(ubuf, ysb, sf_lA, sf_B, sf_C, sf_dt,
                                  sb_lA, sb_B, sb_C, sb_dt);
  run_gemm(0, ysb, F_y, yg, 8192, 256, 128, 128, 1, 0, 0, 0, 256, 0, 1,
           5, nullptr, nullptr, nullptr, 0, xz, 512, 256, stream);            // *gate
  run_gemm(0, yg, F_ol, s1, 8192, 256, 256, 256, 1, 0, 0, 0, 256, 0, 1,
           6, ol_b, nullptr, s0, 256, nullptr, 0, 0, stream);                 // +bias+res

  // ---- Phase 7: seq_out + unpatchify ----
  run_gemm(1, s1, F_seqout, buf2, 8192, 3136, 768, 0, 0, 0, 256, 1024, 3136, 0, 1,
           3, seq_out_b, nullptr, nullptr, 0, nullptr, 0, 0, stream);
  unpatchify<<<EWG, 256, 0, stream>>>(buf2, (float*)d_out);
}
